// MultiHeadAttention_60387240182485
// MI455X (gfx1250) — compile-verified, matrix operand order checked by reading, not run on hardware
//
#include <hip/hip_runtime.h>
#include <hip/hip_bf16.h>
#include <stdint.h>

// ---------------------------------------------------------------------------
// MI455X (gfx1250) fused multi-head attention block, compute-bound design:
//   * all GEMMs on v_wmma_f32_16x16x32_bf16 (wave32, 16x16 tiles)
//   * weights pre-packed to bf16 B-fragment layout once per launch (L2-resident:
//     ~550KB total packed weights vs 192MB L2)
//   * phase1: 1 wave == 1 head; TDM stages input tiles; softmax-over-heads via
//     LDS + barrier; native v_tanh_f32 for activations
//   * phase2: TDM tensor_load_to_lds stages [16x768] ctx tiles, 768->96 WMMA,
//             residual + layernorm
// ---------------------------------------------------------------------------

#define NTOK  131072
#define NHEAD 8
#define HDIM  96

typedef __bf16        v16bf __attribute__((ext_vector_type(16)));
typedef __bf16        v8bf  __attribute__((ext_vector_type(8)));
typedef __bf16        v2bf  __attribute__((ext_vector_type(2)));
typedef float         v8f   __attribute__((ext_vector_type(8)));
typedef float         v2f   __attribute__((ext_vector_type(2)));
typedef unsigned int  u32x4 __attribute__((ext_vector_type(4)));
typedef int           i32x4 __attribute__((ext_vector_type(4)));
typedef int           i32x8 __attribute__((ext_vector_type(8)));

union Frag { v16bf v; u32x4 q[2]; };

// D = A*B + C  (bf16 inputs, f32 accumulate), 16x16x32
__device__ __forceinline__ v8f wmma_bf16(v16bf a, v16bf b, v8f c) {
  return __builtin_amdgcn_wmma_f32_16x16x32_bf16(
      /*neg_a=*/false, a, /*neg_b=*/false, b,
      /*c_mod=*/(short)0, c, /*reuse_a=*/false, /*reuse_b=*/false);
}

// gfx1250 has native v_tanh_f32 (TRANS unit, co-executes with WMMA)
__device__ __forceinline__ float fast_tanh(float x) {
#if __has_builtin(__builtin_amdgcn_tanhf)
  return __builtin_amdgcn_tanhf(x);
#else
  return tanhf(x);
#endif
}

// A-fragment (16x32 bf16, ISA 7.12.2): lane l holds row m=l&15;
// elements 0..7 -> K = kbase + koff + 0..7, elements 8..15 -> +16.
// koff = 8 for lanes 16..31.  Two contiguous 16B ds_load_b128 per lane.
template <int STRIDE>
__device__ __forceinline__ v16bf ldA(const __bf16* s, int lane, int kbase) {
  const __bf16* p =
      s + (size_t)(lane & 15) * STRIDE + kbase + ((lane & 16) ? 8 : 0);
  Frag f;
  f.q[0] = *(const u32x4*)(p);
  f.q[1] = *(const u32x4*)(p + 16);
  return f.v;
}

// B-fragment from pre-packed global memory: frag = 512 bf16, lane-major.
__device__ __forceinline__ v16bf ldB(const __bf16* pack, int lane) {
  Frag f;
  const u32x4* p = (const u32x4*)pack;
  f.q[0] = p[2 * lane];
  f.q[1] = p[2 * lane + 1];
  return f.v;
}

// C/D layout: VGPR v, lane l -> row m = v + ((l&16)?8:0), col n = l&15.
// Single base pointer so stores become immediate-offset ds_store.
__device__ __forceinline__ void stC_f32(float* s, int nbase, int lane, v8f c) {
  float* p = s + (((lane & 16) ? 8 : 0)) * HDIM + nbase + (lane & 15);
#pragma unroll
  for (int v = 0; v < 8; ++v) p[v * HDIM] = c[v];
}
__device__ __forceinline__ void stC_bf16(__bf16* s, int nbase, int lane, v8f c) {
  __bf16* p = s + (((lane & 16) ? 8 : 0)) * HDIM + nbase + (lane & 15);
#pragma unroll
  for (int v = 0; v < 8; ++v) p[v * HDIM] = (__bf16)c[v];
}
__device__ __forceinline__ v8f biasC(const float* b, int nbase, int lane) {
  const float bv = b[nbase + (lane & 15)];
  v8f c = {bv, bv, bv, bv, bv, bv, bv, bv};
  return c;
}

// ---------------------------------------------------------------------------
// Tensor Data Mover: 2D tile load Global->LDS (ISA ch.8 descriptor layout).
//   dim0      : row length in elements (tensor_dim0 == tensor_dim0_stride
//               == tile_dim0, contiguous rows)
//   rows      : tile_dim1
//   tdim1     : tensor_dim1 (rows in whole tensor, for OOB)
//   dsize     : 0=1B 1=2B 2=4B
// ---------------------------------------------------------------------------
__device__ __forceinline__ void tdm_load_tile(const void* gptr, void* lptr,
                                              unsigned dim0, unsigned rows,
                                              unsigned tdim1, unsigned dsize) {
#if __has_builtin(__builtin_amdgcn_tensor_load_to_lds)
  const uint64_t ga = (uint64_t)(uintptr_t)gptr;
  const unsigned lds = (unsigned)(uintptr_t)lptr;
  u32x4 g0 = {1u,                                    // count=1 valid D#
              lds,                                   // lds_addr
              (unsigned)(ga & 0xffffffffu),          // global_addr lo
              (unsigned)((ga >> 32) & 0x01ffffffu) | (2u << 30)};  // hi|type=2
  i32x8 g1;
  g1[0] = (int)(dsize << 16);                        // wg_mask=0, data_size
  g1[1] = (int)(dim0 << 16);                         // tensor_dim0 lo16 @48
  g1[2] = (int)((dim0 >> 16) | ((tdim1 & 0xffffu) << 16));  // d0 hi | d1 lo
  g1[3] = (int)((tdim1 >> 16) | (dim0 << 16));       // d1 hi | tile_dim0
  g1[4] = (int)rows;                                 // tile_dim1 (tile_dim2=0)
  g1[5] = (int)dim0;                                 // tensor_dim0_stride lo
  g1[6] = 0;
  g1[7] = 0;
  i32x4 z4 = {0, 0, 0, 0};
#if __clang_major__ >= 23
  i32x8 z8 = {0, 0, 0, 0, 0, 0, 0, 0};
  __builtin_amdgcn_tensor_load_to_lds(g0, g1, z4, z4, z8, 0);
#else
  __builtin_amdgcn_tensor_load_to_lds(g0, g1, z4, z4, 0);
#endif
  __builtin_amdgcn_s_wait_tensorcnt(0);
#endif
}

// ---------------------------------------------------------------------------
// Kernel 0: pack a weight matrix W[rows][srcCols] (fp32) into bf16 WMMA
// B-fragments:  B[k][n] = W[h*96 + n][k].  One block per fragment.
// ---------------------------------------------------------------------------
__global__ void pack_b_kernel(const float* __restrict__ src,
                              __bf16* __restrict__ dst,
                              int nTiles, int kTiles, int srcCols) {
  const int frag = blockIdx.x;
  const int kt   = frag % kTiles;
  const int t    = frag / kTiles;
  const int nt   = t % nTiles;
  const int h    = t / nTiles;               // head (0 for shared weights)
  const int tid  = threadIdx.x;              // 512 threads
  const int lane = tid >> 4, e = tid & 15;
  const int k = kt * 32 + ((e < 8) ? e : (e + 8)) + ((lane & 16) ? 8 : 0);
  const int n = nt * 16 + (lane & 15);
  dst[(size_t)frag * 512 + tid] =
      (__bf16)src[(size_t)(h * HDIM + n) * srcCols + k];
}

// ---------------------------------------------------------------------------
// Phase 1: one 16-row tile per block, wave w == head w (8 waves / 256 thr).
// ---------------------------------------------------------------------------
__global__ void __launch_bounds__(256)
mha_phase1(const float* __restrict__ inQ, const float* __restrict__ inH,
           const __bf16* __restrict__ pWQ, const __bf16* __restrict__ pWH,
           const __bf16* __restrict__ pWq, const float* wq_b,
           const __bf16* __restrict__ pWh, const float* wh_b,
           const __bf16* __restrict__ pAtt, const float* att_b,
           const __bf16* __restrict__ pUpd, const float* upd_b,
           __bf16* __restrict__ outT) {
  __shared__ __align__(16) __bf16 sQin[16 * HDIM];
  __shared__ __align__(16) __bf16 sHin[16 * HDIM];
  __shared__ __align__(16) __bf16 sQbf[NHEAD][16 * HDIM];  // qp (bf16)
  __shared__ __align__(16) __bf16 sHbf[NHEAD][16 * HDIM];  // hp, later context
  __shared__ __align__(16) float  sHPf[NHEAD][16 * HDIM];  // hp (f32)
  __shared__ __align__(16) float  sEN [NHEAD][16 * HDIM];  // energy / scratch

  const int tid  = threadIdx.x;
  const int lane = tid & 31;
  const int h    = tid >> 5;                 // head id == wave id
  const int r0   = blockIdx.x * 16;

#if __has_builtin(__builtin_amdgcn_tensor_load_to_lds)
  // TDM: stage the two f32 16x96 input tiles into LDS scratch (sEN[0],sEN[1])
  if (h == 0)
    tdm_load_tile(inQ + (size_t)r0 * HDIM, &sEN[0][0], HDIM, 16, NTOK, 2u);
  if (h == 1)
    tdm_load_tile(inH + (size_t)r0 * HDIM, &sEN[1][0], HDIM, 16, NTOK, 2u);
  __syncthreads();
  // convert f32 -> bf16 pairs (v_cvt_pk_bf16_f32)
  for (int i = tid; i < (16 * HDIM) / 2; i += 256) {
    v2f q = ((const v2f*)&sEN[0][0])[i];
    v2f hh = ((const v2f*)&sEN[1][0])[i];
    v2bf qb = {(__bf16)q[0], (__bf16)q[1]};
    v2bf hb = {(__bf16)hh[0], (__bf16)hh[1]};
    ((v2bf*)sQin)[i] = qb;
    ((v2bf*)sHin)[i] = hb;
  }
#else
  for (int i = tid; i < (16 * HDIM) / 2; i += 256) {
    v2f q = ((const v2f*)(inQ + (size_t)r0 * HDIM))[i];
    v2f hh = ((const v2f*)(inH + (size_t)r0 * HDIM))[i];
    v2bf qb = {(__bf16)q[0], (__bf16)q[1]};
    v2bf hb = {(__bf16)hh[0], (__bf16)hh[1]};
    ((v2bf*)sQin)[i] = qb;
    ((v2bf*)sHin)[i] = hb;
  }
#endif
  __syncthreads();

  // sEN[h] doubles as per-head bf16 scratch for the head projections q,hproj
  __bf16* qtmp = (__bf16*)&sEN[h][0];        // 16x96 bf16
  __bf16* htmp = qtmp + 16 * HDIM;           // 16x96 bf16 (fits in 6144B)

  // ---- q = Qin @ W_Q[h]^T, hproj = Hin @ W_H[h]^T -------------------------
#pragma unroll
  for (int nt = 0; nt < 6; ++nt) {
    v8f cq = {0.f, 0.f, 0.f, 0.f, 0.f, 0.f, 0.f, 0.f};
    v8f ch = {0.f, 0.f, 0.f, 0.f, 0.f, 0.f, 0.f, 0.f};
#pragma unroll
    for (int kt = 0; kt < 3; ++kt) {
      const v16bf aq = ldA<HDIM>(sQin, lane, kt * 32);
      const v16bf ah = ldA<HDIM>(sHin, lane, kt * 32);
      cq = wmma_bf16(aq, ldB(pWQ + (size_t)((h * 6 + nt) * 3 + kt) * 512, lane), cq);
      ch = wmma_bf16(ah, ldB(pWH + (size_t)((h * 6 + nt) * 3 + kt) * 512, lane), ch);
    }
    stC_bf16(qtmp, nt * 16, lane, cq);
    stC_bf16(htmp, nt * 16, lane, ch);
  }

  // ---- qp = q @ wq^T + b,  hp = hproj @ wh^T + b --------------------------
#pragma unroll
  for (int nt = 0; nt < 6; ++nt) {
    v8f cq = biasC(wq_b, nt * 16, lane);
    v8f ch = biasC(wh_b, nt * 16, lane);
#pragma unroll
    for (int kt = 0; kt < 3; ++kt) {
      cq = wmma_bf16(ldA<HDIM>(qtmp, lane, kt * 32),
                     ldB(pWq + (size_t)(nt * 3 + kt) * 512, lane), cq);
      ch = wmma_bf16(ldA<HDIM>(htmp, lane, kt * 32),
                     ldB(pWh + (size_t)(nt * 3 + kt) * 512, lane), ch);
    }
    stC_bf16(sQbf[h], nt * 16, lane, cq);
    stC_bf16(sHbf[h], nt * 16, lane, ch);
    stC_f32 (sHPf[h], nt * 16, lane, ch);
  }

  // ---- energy = tanh([hp qp] @ att^T + b)  (K = 192) ----------------------
#pragma unroll
  for (int nt = 0; nt < 6; ++nt) {
    v8f c = biasC(att_b, nt * 16, lane);
#pragma unroll
    for (int kt = 0; kt < 6; ++kt) {
      const __bf16* src = (kt < 3) ? sHbf[h] : sQbf[h];
      const int kb = (kt < 3) ? kt * 32 : (kt - 3) * 32;
      c = wmma_bf16(ldA<HDIM>(src, lane, kb),
                    ldB(pAtt + (size_t)(nt * 6 + kt) * 512, lane), c);
    }
#pragma unroll
    for (int v = 0; v < 8; ++v) c[v] = fast_tanh(c[v]);
    stC_f32(sEN[h], nt * 16, lane, c);   // overwrites qtmp/htmp (done with)
  }
  __syncthreads();

  // ---- softmax over heads, context = attn * hp ----------------------------
  for (int i = lane; i < 16 * HDIM; i += 32) {
    float mx = sEN[0][i];
#pragma unroll
    for (int j = 1; j < NHEAD; ++j) mx = fmaxf(mx, sEN[j][i]);
    float den = 0.f, num = 0.f;
#pragma unroll
    for (int j = 0; j < NHEAD; ++j) {
      const float e = __expf(sEN[j][i] - mx);
      den += e;
      if (j == h) num = e;
    }
    sHbf[h][i] = (__bf16)((num / den) * sHPf[h][i]);  // ctx replaces hp(bf16)
  }
  __syncthreads();   // everyone done reading sEN before it is reused below

  // ---- out = tanh([qp ctx] @ upd^T + b), store [hd][n] transposed ---------
  __bf16* sOut = (__bf16*)&sEN[h][0];           // [96][16] bf16 tile
#pragma unroll
  for (int nt = 0; nt < 6; ++nt) {
    v8f c = biasC(upd_b, nt * 16, lane);
#pragma unroll
    for (int kt = 0; kt < 6; ++kt) {
      const __bf16* src = (kt < 3) ? sQbf[h] : sHbf[h];
      const int kb = (kt < 3) ? kt * 32 : (kt - 3) * 32;
      c = wmma_bf16(ldA<HDIM>(src, lane, kb),
                    ldB(pUpd + (size_t)(nt * 6 + kt) * 512, lane), c);
    }
    // [96][16] layout: lane's 8 outputs are CONTIGUOUS -> one ds_store_b128
    __bf16* p = sOut + (nt * 16 + (lane & 15)) * 16 + ((lane & 16) ? 8 : 0);
    v8bf o;
#pragma unroll
    for (int v = 0; v < 8; ++v) o[v] = (__bf16)fast_tanh(c[v]);
    *(v8bf*)p = o;
  }
  // coalesced 32B stores: lane writes 16 consecutive n for one (head, hd)
#pragma unroll
  for (int it = 0; it < 3; ++it) {
    const int j = it * 32 + lane;
    const u32x4* src = (const u32x4*)&sOut[j * 16];
    u32x4* dst = (u32x4*)(outT + ((size_t)(h * HDIM + j)) * (size_t)NTOK + r0);
    dst[0] = src[0];
    dst[1] = src[1];
  }
}

// ---------------------------------------------------------------------------
// Phase 2: ws == row-major [N][768] bf16 (the reshape is flat re-indexing).
// TDM-load one 16x768 tile to LDS; 6 waves each own one 16-col output tile
// (K=768, 24 WMMA k-steps), + residual input_Q, + layernorm.
// ---------------------------------------------------------------------------
__global__ void __launch_bounds__(192)
mha_phase2(const __bf16* __restrict__ ctx, const __bf16* __restrict__ pFC,
           const float* __restrict__ inQ, const float* __restrict__ ln_g,
           const float* __restrict__ ln_b, float* __restrict__ out) {
  __shared__ __align__(16) __bf16 sCtx[16 * 768];
  __shared__ __align__(16) float  sO[16 * HDIM];

  const int tid  = threadIdx.x;     // 192 threads = 6 waves
  const int lane = tid & 31;
  const int w    = tid >> 5;        // wave == output n-tile
  const int r0   = blockIdx.x * 16;

#if __has_builtin(__builtin_amdgcn_tensor_load_to_lds)
  if (w == 0)
    tdm_load_tile(ctx + (size_t)r0 * 768, &sCtx[0], 768, 16, NTOK, 1u);
#else
  for (int i = tid; i < 16 * 768 / 8; i += 192)
    ((u32x4*)sCtx)[i] = ((const u32x4*)(ctx + (size_t)r0 * 768))[i];
#endif
  __syncthreads();

  // GEMM: o[16x16 tile w] = ctx[16x768] @ FC^T  (24 bf16 WMMA k-steps)
  v8f c = {0.f, 0.f, 0.f, 0.f, 0.f, 0.f, 0.f, 0.f};
#pragma unroll
  for (int kt = 0; kt < 24; ++kt) {
    c = wmma_bf16(ldA<768>(sCtx, lane, kt * 32),
                  ldB(pFC + (size_t)(w * 24 + kt) * 512, lane), c);
  }
  // residual: + input_Q  (base pointer + constant row strides)
  {
    const float* p =
        inQ + (size_t)(r0 + ((lane & 16) ? 8 : 0)) * HDIM + w * 16 + (lane & 15);
#pragma unroll
    for (int v = 0; v < 8; ++v) c[v] += p[v * HDIM];
  }
  stC_f32(sO, w * 16, lane, c);
  __syncthreads();

  // layernorm over last dim (96), one row per thread 0..15
  if (tid < 16) {
    const float* row = &sO[tid * HDIM];
    float mu = 0.f;
#pragma unroll
    for (int i = 0; i < HDIM; ++i) mu += row[i];
    mu *= (1.f / HDIM);
    float var = 0.f;
#pragma unroll
    for (int i = 0; i < HDIM; ++i) {
      const float d = row[i] - mu;
      var += d * d;
    }
    var *= (1.f / HDIM);
    const float inv = rsqrtf(var + 1e-5f);
    for (int i = 0; i < HDIM; ++i)
      out[(size_t)(r0 + tid) * HDIM + i] =
          ln_g[i] * (row[i] - mu) * inv + ln_b[i];
  }
}

// ---------------------------------------------------------------------------
// Launcher
// ---------------------------------------------------------------------------
extern "C" void kernel_launch(void* const* d_in, const int* in_sizes, int n_in,
                              void* d_out, int out_size, void* d_ws,
                              size_t ws_size, hipStream_t stream) {
  const float* inQ   = (const float*)d_in[0];
  const float* inH   = (const float*)d_in[1];
  const float* W_Q   = (const float*)d_in[2];
  const float* W_H   = (const float*)d_in[3];
  const float* wq_w  = (const float*)d_in[4];
  const float* wq_b  = (const float*)d_in[5];
  const float* wh_w  = (const float*)d_in[6];
  const float* wh_b  = (const float*)d_in[7];
  const float* att_w = (const float*)d_in[8];
  const float* att_b = (const float*)d_in[9];
  const float* upd_w = (const float*)d_in[10];
  const float* upd_b = (const float*)d_in[11];
  const float* FC_w  = (const float*)d_in[12];
  const float* ln_g  = (const float*)d_in[13];
  const float* ln_b  = (const float*)d_in[14];
  float* out = (float*)d_out;

  // workspace layout (bf16 elements)
  __bf16* ws = (__bf16*)d_ws;
  const size_t OUTE = (size_t)NHEAD * HDIM * NTOK;  // [head][hd][n] == [N][768]
  __bf16* outT = ws;
  __bf16* pWQ  = outT + OUTE;                        // 8*6*3*512
  __bf16* pWH  = pWQ + (size_t)8 * 6 * 3 * 512;
  __bf16* pWq  = pWH + (size_t)8 * 6 * 3 * 512;      // 6*3*512
  __bf16* pWh  = pWq + (size_t)6 * 3 * 512;
  __bf16* pAtt = pWh + (size_t)6 * 3 * 512;          // 6*6*512
  __bf16* pUpd = pAtt + (size_t)6 * 6 * 512;
  __bf16* pFC  = pUpd + (size_t)6 * 6 * 512;         // 6*24*512

  // weight packing (tiny; all outputs stay L2-resident afterwards)
  pack_b_kernel<<<8 * 6 * 3, 512, 0, stream>>>(W_Q,   pWQ, 6, 3, 96);
  pack_b_kernel<<<8 * 6 * 3, 512, 0, stream>>>(W_H,   pWH, 6, 3, 96);
  pack_b_kernel<<<6 * 3,     512, 0, stream>>>(wq_w,  pWq, 6, 3, 96);
  pack_b_kernel<<<6 * 3,     512, 0, stream>>>(wh_w,  pWh, 6, 3, 96);
  pack_b_kernel<<<6 * 6,     512, 0, stream>>>(att_w, pAtt, 6, 6, 192);
  pack_b_kernel<<<6 * 6,     512, 0, stream>>>(upd_w, pUpd, 6, 6, 192);
  pack_b_kernel<<<6 * 24,    512, 0, stream>>>(FC_w,  pFC, 6, 24, 768);

  mha_phase1<<<NTOK / 16, 256, 0, stream>>>(inQ, inH, pWQ, pWH, pWq, wq_b,
                                            pWh, wh_b, pAtt, att_b, pUpd,
                                            upd_b, outT);
  mha_phase2<<<NTOK / 16, 192, 0, stream>>>(outT, pFC, inQ, ln_g, ln_b, out);
}
